// MotionNN_66451734004270
// MI455X (gfx1250) — compile-verified, matrix operand order checked by reading, not run on hardware
//
#include <hip/hip_runtime.h>
#include <hip/hip_bf16.h>
#include <math.h>
#include <stdint.h>

typedef __bf16 bf16_t;
typedef __bf16 v16bf __attribute__((ext_vector_type(16)));
typedef float  v8f   __attribute__((ext_vector_type(8)));

#define WMMA_BF16(a, b, c) \
  __builtin_amdgcn_wmma_f32_16x16x32_bf16(false, (a), false, (b), (short)0, (c), false, false)

__device__ __forceinline__ v8f vzero8() {
  v8f a;
#pragma unroll
  for (int i = 0; i < 8; i++) a[i] = 0.f;
  return a;
}

// Two contiguous 16B chunks -> one 16x32 bf16 fragment slice for this lane.
// Per ISA layout: lane<16 -> K {0..7, 16..23}; lane>=16 -> K {8..15, 24..31}.
__device__ __forceinline__ v16bf load_frag16(const bf16_t* p0, const bf16_t* p1) {
  union { v16bf v; float4 f[2]; } u;
  u.f[0] = *reinterpret_cast<const float4*>(p0);
  u.f[1] = *reinterpret_cast<const float4*>(p1);
  return u.v;
}
__device__ __forceinline__ v16bf ld_frag_lds(const char* p0, const char* p1) {
  union { v16bf v; float4 f[2]; } u;
  u.f[0] = *reinterpret_cast<const float4*>(p0);
  u.f[1] = *reinterpret_cast<const float4*>(p1);
  return u.v;
}
// Low 32 bits of a flat LDS address = byte offset from the wave's LDS base,
// which is exactly what the async-to-LDS VDST operand wants.
__device__ __forceinline__ unsigned lds_off_of(const void* p) {
  return (unsigned)(uintptr_t)p;
}
__device__ __forceinline__ void async_ld_b128(unsigned ldsoff, const void* gptr) {
  asm volatile("global_load_async_to_lds_b128 %0, %1, off" ::"v"(ldsoff), "v"(gptr)
               : "memory");
}
__device__ __forceinline__ void wait_async0() {
  asm volatile("s_wait_asynccnt 0" ::: "memory");
}

// ---------------------------------------------------------------------------
// Elementwise cast fp32 -> bf16
// ---------------------------------------------------------------------------
__global__ __launch_bounds__(256) void cast_f32_bf16(const float* __restrict__ src,
                                                     bf16_t* __restrict__ dst, int n) {
  int i = blockIdx.x * 256 + threadIdx.x;
  if (i < n) dst[i] = (bf16_t)src[i];
}

// ---------------------------------------------------------------------------
// Cast + transpose weights: src (K,N) fp32 row-major -> dst (N,Kpad) bf16,
// zero-filling k in [K, Kpad).
// ---------------------------------------------------------------------------
__global__ __launch_bounds__(256) void cast_transpose(const float* __restrict__ src,
                                                      bf16_t* __restrict__ dst,
                                                      int K, int N, int Kpad) {
  __shared__ float t[32][33];
  int kb = blockIdx.x * 32, nb = blockIdx.y * 32;
  int tx = threadIdx.x & 31, ty = threadIdx.x >> 5;
#pragma unroll
  for (int i = ty; i < 32; i += 8) {
    int k = kb + i, n = nb + tx;
    t[i][tx] = (k < K && n < N) ? src[(size_t)k * N + n] : 0.f;
  }
  __syncthreads();
#pragma unroll
  for (int i = ty; i < 32; i += 8) {
    int n = nb + i, k = kb + tx;
    if (n < N && k < Kpad) dst[(size_t)n * Kpad + k] = (bf16_t)t[tx][i];
  }
}

// ---------------------------------------------------------------------------
// RMSNorm
// ---------------------------------------------------------------------------
__global__ __launch_bounds__(256) void rmsnorm_k(const float* __restrict__ x,
                                                 const float* __restrict__ w,
                                                 bf16_t* __restrict__ g, int H) {
  __shared__ float red[8];
  int row = blockIdx.x;
  const float* xr = x + (size_t)row * H;
  float s = 0.f;
  for (int i = threadIdx.x; i < H; i += 256) { float v = xr[i]; s += v * v; }
#pragma unroll
  for (int o = 16; o > 0; o >>= 1) s += __shfl_xor(s, o, 32);
  if ((threadIdx.x & 31) == 0) red[threadIdx.x >> 5] = s;
  __syncthreads();
  if (threadIdx.x < 8) {
    float t2 = red[threadIdx.x];
#pragma unroll
    for (int o = 4; o > 0; o >>= 1) t2 += __shfl_xor(t2, o, 8);
    if (threadIdx.x == 0) red[0] = t2;
  }
  __syncthreads();
  float r = rsqrtf(red[0] / (float)H + 1e-6f);
  for (int i = threadIdx.x; i < H; i += 256)
    g[(size_t)row * H + i] = (bf16_t)(xr[i] * r * w[i]);
}

// ---------------------------------------------------------------------------
// RoPE + attention layout prep
// ---------------------------------------------------------------------------
__global__ __launch_bounds__(256) void rope_prep_k(const float* __restrict__ qf,
                                                   const float* __restrict__ kf,
                                                   const float* __restrict__ vf,
                                                   bf16_t* __restrict__ qr,
                                                   bf16_t* __restrict__ kr,
                                                   bf16_t* __restrict__ vt) {
  int idx = blockIdx.x * 256 + threadIdx.x;  // 2^22 total
  int d = idx & 31;
  int s = (idx >> 5) & 511;
  int h = (idx >> 14) & 15;
  int b = idx >> 18;
  size_t src = ((size_t)(b * 512 + s)) * 1024 + h * 64 + d;
  float invf = __powf(10000.f, -(float)d * (1.f / 32.f));
  float ang = (float)s * invf;
  float sn, c;
  __sincosf(ang, &sn, &c);
  size_t bh = (size_t)b * 16 + h;
  size_t drow = (bh * 512 + s) * 64;
  float q1 = qf[src], q2 = qf[src + 32];
  qr[drow + d]      = (bf16_t)(q1 * c - q2 * sn);
  qr[drow + d + 32] = (bf16_t)(q2 * c + q1 * sn);
  float k1 = kf[src], k2 = kf[src + 32];
  kr[drow + d]      = (bf16_t)(k1 * c - k2 * sn);
  kr[drow + d + 32] = (bf16_t)(k2 * c + k1 * sn);
  vt[(bh * 64 + d) * 512 + s]      = (bf16_t)vf[src];
  vt[(bh * 64 + d + 32) * 512 + s] = (bf16_t)vf[src + 32];
}

// ---------------------------------------------------------------------------
// Async-staged WMMA GEMM: C(M,N) = A_bf16(M,K) @ Bt_bf16(N,K)^T + bias.
// Block 256 thr = 8 waves (2x4), block tile 128x256, wave tile 64x64,
// K-step 64, double-buffered global_load_async_to_lds_b128 staging.
// LDS row stride 144B (36 dwords; gcd(36,64)=4) -> conflict-free ds_load_b128.
// EPI 0: Cf = v ; EPI 1: Hres += v*rs ; EPI 2: Cb = silu(U)*v (bf16, 0 pad)
// ---------------------------------------------------------------------------
#define GSTRIDE 144
#define ABYTES (128 * GSTRIDE)  // 18432
#define BBYTES (256 * GSTRIDE)  // 36864
#define BUFB (ABYTES + BBYTES)  // 55296

__device__ __forceinline__ void stage_tile(const bf16_t* __restrict__ A,
                                           const bf16_t* __restrict__ Bt,
                                           char* smem, int buf, int bm, int bn,
                                           int kk, int K, int Nreal, int tid) {
  char* dstA = smem + buf * BUFB;
  char* dstB = dstA + ABYTES;
#pragma unroll
  for (int j = 0; j < 4; j++) {  // A: 128 rows x 8 chunks = 1024 chunks
    int c = j * 256 + tid;
    int row = c >> 3, sub = c & 7;
    const bf16_t* g = A + (size_t)(bm + row) * K + kk + sub * 8;
    async_ld_b128(lds_off_of(dstA + row * GSTRIDE + sub * 16), g);
  }
#pragma unroll
  for (int j = 0; j < 8; j++) {  // B: 256 rows x 8 chunks = 2048 chunks
    int c = j * 256 + tid;
    int row = c >> 3, sub = c & 7;
    int nr = bn + row;
    if (nr >= Nreal) nr = Nreal - 1;  // clamp padded rows; stores fix pads
    const bf16_t* g = Bt + (size_t)nr * K + kk + sub * 8;
    async_ld_b128(lds_off_of(dstB + row * GSTRIDE + sub * 16), g);
  }
}

template <int EPI>
__global__ __launch_bounds__(256) void gemm_async_k(const bf16_t* __restrict__ A,
                                                    const bf16_t* __restrict__ Bt,
                                                    const float* __restrict__ bias,
                                                    float* __restrict__ Cf,
                                                    bf16_t* __restrict__ Cb,
                                                    const float* __restrict__ U,
                                                    float* __restrict__ Hres,
                                                    int M, int N, int K, int Nreal,
                                                    float rs) {
  __shared__ __align__(16) char smem[2 * BUFB];  // 108 KB
  const int tid = threadIdx.x, lane = tid & 31, wave = tid >> 5;
  const int wm = wave >> 2, wn = wave & 3;
  const int bm = blockIdx.y * 128, bn = blockIdx.x * 256;
  const int m0w = wm * 64, n0w = wn * 64;
  const int half = lane >> 4, l15 = lane & 15;

  v8f acc[4][4];
#pragma unroll
  for (int mi = 0; mi < 4; mi++)
#pragma unroll
    for (int ni = 0; ni < 4; ni++) acc[mi][ni] = vzero8();

  const int ktiles = K >> 6;
  stage_tile(A, Bt, smem, 0, bm, bn, 0, K, Nreal, tid);

  for (int t = 0; t < ktiles; t++) {
    wait_async0();     // own slice of tile t landed in LDS
    __syncthreads();   // everyone's slice landed; prior buffer free for reuse
    if (t + 1 < ktiles)
      stage_tile(A, Bt, smem, (t + 1) & 1, bm, bn, (t + 1) * 64, K, Nreal, tid);
    const char* base = smem + (t & 1) * BUFB;
#pragma unroll
    for (int ks = 0; ks < 2; ks++) {
      v16bf af[4], bfr[4];
#pragma unroll
      for (int mi = 0; mi < 4; mi++) {
        const char* pa = base + (size_t)(m0w + mi * 16 + l15) * GSTRIDE + ks * 64 + half * 16;
        af[mi] = ld_frag_lds(pa, pa + 32);
      }
#pragma unroll
      for (int ni = 0; ni < 4; ni++) {
        const char* pb = base + ABYTES + (size_t)(n0w + ni * 16 + l15) * GSTRIDE + ks * 64 + half * 16;
        bfr[ni] = ld_frag_lds(pb, pb + 32);
      }
#pragma unroll
      for (int mi = 0; mi < 4; mi++)
#pragma unroll
        for (int ni = 0; ni < 4; ni++)
          acc[mi][ni] = WMMA_BF16(af[mi], bfr[ni], acc[mi][ni]);
    }
  }

#pragma unroll
  for (int mi = 0; mi < 4; mi++) {
#pragma unroll
    for (int ni = 0; ni < 4; ni++) {
      int colg = bn + n0w + ni * 16 + l15;
      int colc = (colg < Nreal) ? colg : (Nreal - 1);
      float bv = bias ? bias[colc] : 0.f;
#pragma unroll
      for (int i = 0; i < 8; i++) {
        int r = bm + m0w + mi * 16 + i + half * 8;  // lanes>=16 hold M+8
        size_t idx = (size_t)r * N + colg;
        float v = acc[mi][ni][i] + bv;
        if (EPI == 0) {
          Cf[idx] = v;
        } else if (EPI == 1) {
          Hres[idx] += v * rs;
        } else {
          float outv = 0.f;
          if (colg < Nreal) {
            float uu = U[idx];
            outv = (uu / (1.f + __expf(-uu))) * v;  // silu(u) * t
          }
          Cb[idx] = (bf16_t)outv;
        }
      }
    }
  }
}

// ---------------------------------------------------------------------------
// Direct-load WMMA GEMM (kept for the small N=128 output projection).
// Block tile 128x128, wave tile 64x32.
// ---------------------------------------------------------------------------
template <int EPI>
__global__ __launch_bounds__(256) void gemm_k(const bf16_t* __restrict__ A,
                                              const bf16_t* __restrict__ Bt,
                                              const float* __restrict__ bias,
                                              float* __restrict__ Cf,
                                              bf16_t* __restrict__ Cb,
                                              const float* __restrict__ U,
                                              float* __restrict__ Hres,
                                              int M, int N, int K, int Nreal, float rs) {
  const int tid = threadIdx.x, lane = tid & 31, wave = tid >> 5;
  const int wm = wave >> 2, wn = wave & 3;
  const int m0 = blockIdx.y * 128 + wm * 64;
  const int n0 = blockIdx.x * 128 + wn * 32;
  const int half = lane >> 4, l15 = lane & 15;

  v8f acc[4][2];
#pragma unroll
  for (int mi = 0; mi < 4; mi++)
#pragma unroll
    for (int ni = 0; ni < 2; ni++) acc[mi][ni] = vzero8();

  const bf16_t* Arow[4];
#pragma unroll
  for (int mi = 0; mi < 4; mi++) Arow[mi] = A + (size_t)(m0 + mi * 16 + l15) * K;
  const bf16_t* Brow[2];
#pragma unroll
  for (int ni = 0; ni < 2; ni++) {
    int nr = n0 + ni * 16 + l15;
    if (nr >= Nreal) nr = Nreal - 1;
    Brow[ni] = Bt + (size_t)nr * K;
  }

  for (int kk = 0; kk < K; kk += 32) {
    v16bf b0 = load_frag16(Brow[0] + kk + half * 8, Brow[0] + kk + 16 + half * 8);
    v16bf b1 = load_frag16(Brow[1] + kk + half * 8, Brow[1] + kk + 16 + half * 8);
#pragma unroll
    for (int mi = 0; mi < 4; mi++) {
      v16bf a = load_frag16(Arow[mi] + kk + half * 8, Arow[mi] + kk + 16 + half * 8);
      acc[mi][0] = WMMA_BF16(a, b0, acc[mi][0]);
      acc[mi][1] = WMMA_BF16(a, b1, acc[mi][1]);
    }
  }

#pragma unroll
  for (int mi = 0; mi < 4; mi++) {
#pragma unroll
    for (int ni = 0; ni < 2; ni++) {
      int colg = n0 + ni * 16 + l15;
      int colc = (colg < Nreal) ? colg : (Nreal - 1);
      float bv = bias ? bias[colc] : 0.f;
#pragma unroll
      for (int i = 0; i < 8; i++) {
        int r = m0 + mi * 16 + i + half * 8;
        size_t idx = (size_t)r * N + colg;
        float v = acc[mi][ni][i] + bv;
        if (EPI == 0) {
          Cf[idx] = v;
        } else if (EPI == 1) {
          Hres[idx] += v * rs;
        } else {
          float outv = 0.f;
          if (colg < Nreal) {
            float uu = U[idx];
            outv = (uu / (1.f + __expf(-uu))) * v;
          }
          Cb[idx] = (bf16_t)outv;
        }
      }
    }
  }
}

// ---------------------------------------------------------------------------
// Attention (full softmax per reference; no mask). 64 query rows per block.
// ---------------------------------------------------------------------------
#define AS 512
#define AHD 64
__global__ __launch_bounds__(256) void attention_k(const bf16_t* __restrict__ qr,
                                                   const bf16_t* __restrict__ kr,
                                                   const bf16_t* __restrict__ vt,
                                                   bf16_t* __restrict__ o) {
  __shared__ float sc[64 * AS];
  __shared__ float op[64 * AHD];
  __shared__ float red[64 * 4];
  __shared__ float rowm[64];
  __shared__ float rowsum[64];

  const int bh = blockIdx.y;
  const int q0 = blockIdx.x * 64;
  const int tid = threadIdx.x, lane = tid & 31, wave = tid >> 5;
  const int wm = wave & 3, wh = wave >> 2;
  const int b = bh >> 4, hh = bh & 15;
  const int half = lane >> 4, l15 = lane & 15;

  const bf16_t* Q  = qr + (size_t)bh * AS * AHD;
  const bf16_t* Kr = kr + (size_t)bh * AS * AHD;
  const bf16_t* Vt = vt + (size_t)bh * AHD * AS;

  {  // scores
    v8f nf[16];
#pragma unroll
    for (int i = 0; i < 16; i++) nf[i] = vzero8();
    int mrow = q0 + wm * 16 + l15;
#pragma unroll
    for (int ks = 0; ks < 2; ks++) {
      int kk = ks * 32;
      v16bf a = load_frag16(Q + (size_t)mrow * AHD + kk + half * 8,
                            Q + (size_t)mrow * AHD + kk + 16 + half * 8);
#pragma unroll
      for (int ni = 0; ni < 16; ni++) {
        int nrow = wh * 256 + ni * 16 + l15;
        v16bf bb = load_frag16(Kr + (size_t)nrow * AHD + kk + half * 8,
                               Kr + (size_t)nrow * AHD + kk + 16 + half * 8);
        nf[ni] = WMMA_BF16(a, bb, nf[ni]);
      }
    }
    const float scal = 0.125f;
#pragma unroll
    for (int ni = 0; ni < 16; ni++) {
      int col = wh * 256 + ni * 16 + l15;
#pragma unroll
      for (int i = 0; i < 8; i++) {
        int r = wm * 16 + i + half * 8;
        sc[r * AS + col] = nf[ni][i] * scal;
      }
    }
  }
  __syncthreads();

  {  // softmax, 4 threads per row
    int r = tid >> 2, p = tid & 3;
    float* row = sc + r * AS + p * 128;
    float m = -1e30f;
    for (int i = 0; i < 128; i++) m = fmaxf(m, row[i]);
    red[r * 4 + p] = m;
    __syncthreads();
    if (p == 0)
      rowm[r] = fmaxf(fmaxf(red[r * 4], red[r * 4 + 1]),
                      fmaxf(red[r * 4 + 2], red[r * 4 + 3]));
    __syncthreads();
    float mm = rowm[r];
    float s = 0.f;
    for (int i = 0; i < 128; i++) { float e = __expf(row[i] - mm); row[i] = e; s += e; }
    red[r * 4 + p] = s;
    __syncthreads();
    if (p == 0)
      rowsum[r] = 1.f / (red[r * 4] + red[r * 4 + 1] + red[r * 4 + 2] + red[r * 4 + 3]);
    __syncthreads();
    float inv = rowsum[r];
    for (int i = 0; i < 128; i++) row[i] *= inv;
  }
  __syncthreads();

  {  // O = P @ V
    v8f of[4];
#pragma unroll
    for (int i = 0; i < 4; i++) of[i] = vzero8();
    int mr = wm * 16 + l15;
    for (int ks = 0; ks < 8; ks++) {
      int kb = wh * 256 + ks * 32;
      union { v16bf v; bf16_t e[16]; } ua;
      const float* pr = sc + mr * AS + kb + half * 8;
#pragma unroll
      for (int j = 0; j < 8; j++) ua.e[j] = (bf16_t)pr[j];
#pragma unroll
      for (int j = 0; j < 8; j++) ua.e[8 + j] = (bf16_t)pr[16 + j];
#pragma unroll
      for (int ni = 0; ni < 4; ni++) {
        int d = ni * 16 + l15;
        v16bf bb = load_frag16(Vt + (size_t)d * AS + kb + half * 8,
                               Vt + (size_t)d * AS + kb + 16 + half * 8);
        of[ni] = WMMA_BF16(ua.v, bb, of[ni]);
      }
    }
    if (wh == 1) {
#pragma unroll
      for (int ni = 0; ni < 4; ni++) {
        int col = ni * 16 + l15;
#pragma unroll
        for (int i = 0; i < 8; i++) {
          int r = wm * 16 + i + half * 8;
          op[r * AHD + col] = of[ni][i];
        }
      }
    }
    __syncthreads();
    if (wh == 0) {
#pragma unroll
      for (int ni = 0; ni < 4; ni++) {
        int col = ni * 16 + l15;
#pragma unroll
        for (int i = 0; i < 8; i++) {
          int r = wm * 16 + i + half * 8;
          float v = of[ni][i] + op[r * AHD + col];
          o[((size_t)(b * AS + q0 + r)) * 1024 + hh * 64 + col] = (bf16_t)v;
        }
      }
    }
  }
}

// ---------------------------------------------------------------------------
// Probe (not launched): cluster/tensor builtins kept as validated references.
// ---------------------------------------------------------------------------
__global__ void cdna5_probe_async(const float* __restrict__ src, float* __restrict__ dst) {
  __shared__ float lbuf[256];
  unsigned lofs = threadIdx.x * 4u;
  const float* p = src + threadIdx.x;
  asm volatile("global_load_async_to_lds_b32 %0, %1, off" ::"v"(lofs), "v"(p) : "memory");
  asm volatile("s_wait_asynccnt 0" ::: "memory");
  __syncthreads();
  int cid = __builtin_amdgcn_cluster_id_x();
  __builtin_amdgcn_s_cluster_barrier();
  __builtin_amdgcn_s_wait_tensorcnt(0);
  dst[threadIdx.x] = lbuf[threadIdx.x] + (float)cid;
}

// ---------------------------------------------------------------------------
// Host launcher
// ---------------------------------------------------------------------------
extern "C" void kernel_launch(void* const* d_in, const int* in_sizes, int n_in,
                              void* d_out, int out_size, void* d_ws, size_t ws_size,
                              hipStream_t stream) {
  (void)in_sizes; (void)n_in; (void)out_size; (void)ws_size;
  const int Dd = 128, Hh = 1024, Aa = 2730, Apad = 2816;
  const int Mtok = 8192;
  const float RSc = 0.40824829046386307f;  // 1/sqrt(6)

  const float* x      = (const float*)d_in[0];
  const float* in_w   = (const float*)d_in[1];
  const float* in_b   = (const float*)d_in[2];
  const float* norm1  = (const float*)d_in[3];
  const float* norm2  = (const float*)d_in[4];
  const float* wq     = (const float*)d_in[5];
  const float* bq     = (const float*)d_in[6];
  const float* wk     = (const float*)d_in[7];
  const float* bk     = (const float*)d_in[8];
  const float* wv     = (const float*)d_in[9];
  const float* bv     = (const float*)d_in[10];
  const float* wo     = (const float*)d_in[11];
  const float* bo     = (const float*)d_in[12];
  const float* w1     = (const float*)d_in[13];
  const float* b1     = (const float*)d_in[14];
  const float* w3     = (const float*)d_in[15];
  const float* b3     = (const float*)d_in[16];
  const float* w2     = (const float*)d_in[17];
  const float* b2     = (const float*)d_in[18];
  const float* onormw = (const float*)d_in[19];
  const float* out_w  = (const float*)d_in[20];
  const float* out_b  = (const float*)d_in[21];

  char* ws = (char*)d_ws;
  size_t off = 0;
  auto take = [&](size_t bytes) -> char* {
    off = (off + 255) & ~(size_t)255;
    char* p = ws + off;
    off += bytes;
    return p;
  };

  bf16_t* wt_in = (bf16_t*)take((size_t)Hh * Dd * 2);
  bf16_t *wtq[6], *wtk[6], *wtv[6], *wto[6], *wt1[6], *wt3[6], *wt2[6];
  for (int l = 0; l < 6; l++) {
    wtq[l] = (bf16_t*)take((size_t)Hh * Hh * 2);
    wtk[l] = (bf16_t*)take((size_t)Hh * Hh * 2);
    wtv[l] = (bf16_t*)take((size_t)Hh * Hh * 2);
    wto[l] = (bf16_t*)take((size_t)Hh * Hh * 2);
    wt1[l] = (bf16_t*)take((size_t)Aa * Hh * 2);
    wt3[l] = (bf16_t*)take((size_t)Aa * Hh * 2);
    wt2[l] = (bf16_t*)take((size_t)Hh * Apad * 2);
  }
  bf16_t* wt_out = (bf16_t*)take((size_t)Dd * Hh * 2);
  bf16_t* xb   = (bf16_t*)take((size_t)Mtok * Dd * 2);
  float*  hbuf = (float*)take((size_t)Mtok * Hh * 4);
  bf16_t* gbuf = (bf16_t*)take((size_t)Mtok * Hh * 2);
  float*  big0 = (float*)take((size_t)Mtok * Hh * 4 * 3);  // qf|kf|vf; reused as ubuf
  float* qf = big0;
  float* kf = big0 + (size_t)Mtok * Hh;
  float* vf = big0 + 2 * (size_t)Mtok * Hh;
  float* ubuf = big0;
  bf16_t* big1 = (bf16_t*)take((size_t)48 * 1024 * 1024);  // qr|kr|vt; reused as ff
  bf16_t* qr = big1;
  bf16_t* kr = big1 + (size_t)Mtok * Hh;
  bf16_t* vt = big1 + 2 * (size_t)Mtok * Hh;
  bf16_t* ff = big1;
  bf16_t* obuf = (bf16_t*)take((size_t)Mtok * Hh * 2);

  dim3 blk(256);
  cast_f32_bf16<<<(Mtok * Dd + 255) / 256, blk, 0, stream>>>(x, xb, Mtok * Dd);

  auto ct = [&](const float* src, bf16_t* dst, int K, int N, int Kpad) {
    dim3 g((Kpad + 31) / 32, (N + 31) / 32);
    cast_transpose<<<g, blk, 0, stream>>>(src, dst, K, N, Kpad);
  };
  ct(in_w, wt_in, Dd, Hh, Dd);
  for (int l = 0; l < 6; l++) {
    ct(wq + (size_t)l * Hh * Hh, wtq[l], Hh, Hh, Hh);
    ct(wk + (size_t)l * Hh * Hh, wtk[l], Hh, Hh, Hh);
    ct(wv + (size_t)l * Hh * Hh, wtv[l], Hh, Hh, Hh);
    ct(wo + (size_t)l * Hh * Hh, wto[l], Hh, Hh, Hh);
    ct(w1 + (size_t)l * Hh * Aa, wt1[l], Hh, Aa, Hh);
    ct(w3 + (size_t)l * Hh * Aa, wt3[l], Hh, Aa, Hh);
    ct(w2 + (size_t)l * Aa * Hh, wt2[l], Aa, Hh, Apad);
  }
  ct(out_w, wt_out, Hh, Dd, Hh);

  // input projection: h = x @ in_w + in_b
  gemm_async_k<0><<<dim3(Hh / 256, Mtok / 128), blk, 0, stream>>>(
      xb, wt_in, in_b, hbuf, nullptr, nullptr, nullptr, Mtok, Hh, Dd, Hh, 0.f);

  for (int l = 0; l < 6; l++) {
    rmsnorm_k<<<Mtok, blk, 0, stream>>>(hbuf, norm1 + (size_t)l * Hh, gbuf, Hh);
    gemm_async_k<0><<<dim3(4, 64), blk, 0, stream>>>(gbuf, wtq[l], bq + (size_t)l * Hh, qf,
                                                     nullptr, nullptr, nullptr, Mtok, Hh, Hh, Hh, 0.f);
    gemm_async_k<0><<<dim3(4, 64), blk, 0, stream>>>(gbuf, wtk[l], bk + (size_t)l * Hh, kf,
                                                     nullptr, nullptr, nullptr, Mtok, Hh, Hh, Hh, 0.f);
    gemm_async_k<0><<<dim3(4, 64), blk, 0, stream>>>(gbuf, wtv[l], bv + (size_t)l * Hh, vf,
                                                     nullptr, nullptr, nullptr, Mtok, Hh, Hh, Hh, 0.f);
    rope_prep_k<<<16384, blk, 0, stream>>>(qf, kf, vf, qr, kr, vt);
    attention_k<<<dim3(8, 256), blk, 0, stream>>>(qr, kr, vt, obuf);
    gemm_async_k<1><<<dim3(4, 64), blk, 0, stream>>>(obuf, wto[l], bo + (size_t)l * Hh, nullptr,
                                                     nullptr, nullptr, hbuf, Mtok, Hh, Hh, Hh, RSc);
    rmsnorm_k<<<Mtok, blk, 0, stream>>>(hbuf, norm2 + (size_t)l * Hh, gbuf, Hh);
    gemm_async_k<0><<<dim3(Apad / 256, 64), blk, 0, stream>>>(
        gbuf, wt1[l], b1 + (size_t)l * Aa, ubuf, nullptr, nullptr, nullptr, Mtok, Apad, Hh, Aa, 0.f);
    gemm_async_k<2><<<dim3(Apad / 256, 64), blk, 0, stream>>>(
        gbuf, wt3[l], b3 + (size_t)l * Aa, nullptr, ff, ubuf, nullptr, Mtok, Apad, Hh, Aa, 0.f);
    gemm_async_k<1><<<dim3(4, 64), blk, 0, stream>>>(ff, wt2[l], b2 + (size_t)l * Hh, nullptr,
                                                     nullptr, nullptr, hbuf, Mtok, Hh, Apad, Hh, RSc);
  }
  rmsnorm_k<<<Mtok, blk, 0, stream>>>(hbuf, onormw, gbuf, Hh);
  gemm_k<0><<<dim3(1, 64), blk, 0, stream>>>(gbuf, wt_out, out_b, (float*)d_out,
                                             nullptr, nullptr, nullptr, Mtok, Dd, Hh, Dd, 0.f);
}
